// RWKV_TimeMix_RWKV5_51539607552631
// MI455X (gfx1250) — matrix operand order repaired and checked by reading, not compiled, not run on hardware
//
#include <hip/hip_runtime.h>

// ---------------- problem constants ----------------
constexpr int B_  = 8;
constexpr int T_  = 2048;
constexpr int C_  = 2048;
constexpr int HS_ = 64;
constexpr int H_  = C_ / HS_;            // 32
constexpr int GM  = B_ * T_;             // 16384 rows (tokens)
constexpr int GN  = C_;                  // 2048
constexpr int GK  = C_;                  // 2048
constexpr float DIVISOR_ = 8.0f;

// ---------------- WMMA types ----------------
typedef __attribute__((ext_vector_type(16))) __bf16 v16bf;
typedef __attribute__((ext_vector_type(8)))  __bf16 v8bf;
typedef __attribute__((ext_vector_type(8)))  float  v8f;

union Frag { v16bf v; v8bf h[2]; };

// ---------------- GEMM tiling ----------------
constexpr int BM = 128, BN = 128, BK = 32;
constexpr int LDA = 40;   // padded row length (halves) for A tile  [M][K]
constexpr int LDB = 40;   // padded row length (halves) for B tile  [N][K] (transposed)
constexpr int NK = GK / BK;  // 64 K-steps

// =====================================================================
// 1) token-shift + lerp, f32 -> bf16 activations (xk, xv, xr, xg)
// =====================================================================
__global__ __launch_bounds__(256) void prep_kernel(
    const float* __restrict__ x,
    const float* __restrict__ tmk, const float* __restrict__ tmv,
    const float* __restrict__ tmr, const float* __restrict__ tmg,
    __bf16* __restrict__ xk, __bf16* __restrict__ xv,
    __bf16* __restrict__ xr, __bf16* __restrict__ xg)
{
    size_t i = (size_t)blockIdx.x * blockDim.x + threadIdx.x;   // over B*T*C
    int c = (int)(i % C_);
    size_t bt = i / C_;
    int t = (int)(bt % T_);
    float xc = x[i];
    float xp = (t == 0) ? 0.0f : x[i - C_];
    float mk = tmk[c], mv = tmv[c], mr = tmr[c], mg = tmg[c];
    xk[i] = (__bf16)(xc * mk + xp * (1.0f - mk));
    xv[i] = (__bf16)(xc * mv + xp * (1.0f - mv));
    xr[i] = (__bf16)(xc * mr + xp * (1.0f - mr));
    xg[i] = (__bf16)(xc * mg + xp * (1.0f - mg));
}

// =====================================================================
// 2) f32 -> bf16 weight conversion
// =====================================================================
__global__ __launch_bounds__(256) void f32_to_bf16_kernel(
    const float* __restrict__ in, __bf16* __restrict__ out, int n)
{
    int i = blockIdx.x * blockDim.x + threadIdx.x;
    if (i < n) out[i] = (__bf16)in[i];
}

// =====================================================================
// 3) bf16 WMMA GEMM:  C(f32) = A(bf16, MxK row-major) * W(bf16, KxN row-major)
//    block tile 128x128, 8 waves -> each wave 64x32 (4x2 grid of 16x16x32 WMMA)
//    A tile: async global->LDS copy (no VGPR staging, ASYNCcnt tracked)
//    B tile: register-staged with transpose into [N][K]
//    A fragments double-buffered so LDS latency hides behind WMMAs
// =====================================================================
__global__ __launch_bounds__(256, 1) void gemm_bf16_wmma(
    const __bf16* __restrict__ A, const __bf16* __restrict__ Bw,
    float* __restrict__ Cout)
{
    __shared__ __bf16 sA[2][BM * LDA];   // [M][K]
    __shared__ __bf16 sB[2][BN * LDB];   // [N][K]  (transposed weight tile)

    const int tid   = threadIdx.x;
    const int wave  = tid >> 5;
    const int lane  = tid & 31;
    const int group = lane >> 4;     // 0: lanes 0-15, 1: lanes 16-31
    const int l16   = lane & 15;
    const int wm    = wave & 1;      // 64-row band within 128
    const int wn    = wave >> 1;     // 32-col band within 128

    const size_t rowBase = (size_t)blockIdx.y * BM;
    const size_t colBase = (size_t)blockIdx.x * BN;

    v8f acc[4][2];
    #pragma unroll
    for (int mi = 0; mi < 4; ++mi)
        #pragma unroll
        for (int ni = 0; ni < 2; ++ni)
            #pragma unroll
            for (int e = 0; e < 8; ++e) acc[mi][ni][e] = 0.0f;

    uint4 rb[2];

    // --- async A-tile copy: global -> LDS, no VGPR staging (ASYNCcnt) ---
    auto asyncA = [&](int kt, int buf) {
        #pragma unroll
        for (int i = 0; i < 2; ++i) {
            int c = tid + 256 * i;                 // 512 chunks of 8 halves
            int row = c >> 2, kc = c & 3;
            const __bf16* gp = A + (rowBase + row) * (size_t)GK
                                 + (size_t)kt * BK + kc * 8;
            unsigned int laddr =
                (unsigned int)(uintptr_t)&sA[buf][row * LDA + kc * 8];
            unsigned long long gaddr = (unsigned long long)(uintptr_t)gp;
            asm volatile("global_load_async_to_lds_b128 %0, %1, off"
                         :: "v"(laddr), "v"(gaddr)
                         : "memory");
        }
    };
    auto loadB = [&](int kt) {
        #pragma unroll
        for (int i = 0; i < 2; ++i) {
            int c = tid + 256 * i;
            int kk = c >> 4, nc = c & 15;
            rb[i] = *(const uint4*)(Bw + (size_t)((size_t)kt * BK + kk) * GN
                                       + colBase + nc * 8);
        }
    };
    auto stageB = [&](int buf) {
        #pragma unroll
        for (int i = 0; i < 2; ++i) {
            int c = tid + 256 * i;
            int kk = c >> 4, nc = c & 15;
            const __bf16* p = (const __bf16*)&rb[i];
            #pragma unroll
            for (int j = 0; j < 8; ++j)
                sB[buf][(nc * 8 + j) * LDB + kk] = p[j];   // transpose into [N][K]
        }
    };
    auto compute = [&](int buf) {
        // B fragments: lane holds column N=l16; halves 0..7 -> K = 8*group + h,
        // halves 8..15 -> K = 16 + 8*group + (h-8)
        Frag fb[2];
        #pragma unroll
        for (int ni = 0; ni < 2; ++ni) {
            const __bf16* brow = &sB[buf][(wn * 32 + ni * 16 + l16) * LDB];
            fb[ni].h[0] = *(const v8bf*)(brow + 8 * group);
            fb[ni].h[1] = *(const v8bf*)(brow + 16 + 8 * group);
        }
        // A fragments double-buffered: loads for mi+1 issue before WMMAs of mi.
        // lane holds row M=l16; halves 0..7 -> K = 8*group + h,
        // halves 8..15 -> K = 16 + 8*group + (h-8)
        Frag fa0, fa1;
        auto loadFa = [&](int mi, Frag& f) {
            const __bf16* arow = &sA[buf][(wm * 64 + mi * 16 + l16) * LDA];
            f.h[0] = *(const v8bf*)(arow + 8 * group);
            f.h[1] = *(const v8bf*)(arow + 16 + 8 * group);
        };
        loadFa(0, fa0);
        #pragma unroll
        for (int mi = 0; mi < 4; ++mi) {
            Frag& cur = (mi & 1) ? fa1 : fa0;
            Frag& nxt = (mi & 1) ? fa0 : fa1;
            if (mi + 1 < 4) loadFa(mi + 1, nxt);      // prefetch next A frag
            acc[mi][0] = __builtin_amdgcn_wmma_f32_16x16x32_bf16(
                false, cur.v, false, fb[0].v, (short)0, acc[mi][0], false, false);
            acc[mi][1] = __builtin_amdgcn_wmma_f32_16x16x32_bf16(
                false, cur.v, false, fb[1].v, (short)0, acc[mi][1], false, false);
        }
    };

    asyncA(0, 0);
    loadB(0);
    stageB(0);
    asm volatile("s_wait_asynccnt 0x0" ::: "memory");
    __syncthreads();

    for (int kt = 0; kt < NK; ++kt) {
        const int cur = kt & 1;
        if (kt + 1 < NK) {
            asyncA(kt + 1, 1 - cur);    // async copy overlaps WMMA compute
            loadB(kt + 1);              // B staged via regs (transpose)
        }
        if (kt + 2 < NK) {              // near-cache prefetch for B two steps out
            __builtin_prefetch(Bw + (size_t)((size_t)(kt + 2) * BK + (tid >> 4)) * GN
                                  + colBase, 0, 3);
        }
        compute(cur);
        if (kt + 1 < NK) stageB(1 - cur);
        asm volatile("s_wait_asynccnt 0x0" ::: "memory");
        __syncthreads();
    }

    // epilogue: C/D layout — VGPR j of lane: M = j + 8*group, N = l16
    #pragma unroll
    for (int mi = 0; mi < 4; ++mi)
        #pragma unroll
        for (int ni = 0; ni < 2; ++ni)
            #pragma unroll
            for (int j = 0; j < 8; ++j) {
                size_t r = rowBase + wm * 64 + mi * 16 + j + 8 * group;
                size_t c = colBase + wn * 32 + ni * 16 + l16;
                Cout[r * (size_t)GN + c] = acc[mi][ni][j];
            }
}

// =====================================================================
// 4) WKV5 recurrence: one 64-thread block per (b,h); S[64] in registers
// =====================================================================
__global__ __launch_bounds__(64) void wkv5_kernel(
    const float* __restrict__ r, const float* __restrict__ k,
    const float* __restrict__ v,
    const float* __restrict__ decay, const float* __restrict__ faaaa,
    float* __restrict__ y)
{
    const int bh = blockIdx.x;
    const int b  = bh / H_;
    const int h  = bh % H_;
    const int j  = threadIdx.x;        // state column / v index

    __shared__ float rs[HS_], ks[HS_], ws[HS_], us[HS_];
    ws[j] = __expf(-__expf(decay[h * HS_ + j]));
    us[j] = faaaa[h * HS_ + j];

    float S[HS_];
    #pragma unroll
    for (int i = 0; i < HS_; ++i) S[i] = 0.0f;
    __syncthreads();

    const size_t base0 = (size_t)b * T_ * C_ + (size_t)h * HS_;
    for (int t = 0; t < T_; ++t) {
        const size_t base = base0 + (size_t)t * C_;
        rs[j] = r[base + j];
        ks[j] = k[base + j];
        const float vj = v[base + j];
        __syncthreads();
        float yj = 0.0f;
        #pragma unroll
        for (int i = 0; i < HS_; ++i) {
            const float kv = ks[i] * vj;
            yj += rs[i] * (S[i] + us[i] * kv);
            S[i] = ws[i] * S[i] + kv;
        }
        y[base + j] = yj;
        __syncthreads();
    }
}

// =====================================================================
// 5) per-head groupnorm + SiLU gate -> bf16 GEMM operand
// =====================================================================
__global__ __launch_bounds__(64) void norm_gate_kernel(
    const float* __restrict__ y, const float* __restrict__ gpre,
    const float* __restrict__ ln_w, const float* __restrict__ ln_b,
    __bf16* __restrict__ aout)
{
    const int idx = blockIdx.x;             // over B*T*H
    const int h   = idx % H_;
    const size_t base = (size_t)idx * HS_;  // contiguous head slice in (B,T,C)
    const int i = threadIdx.x;

    const float yh = y[base + i] * (1.0f / DIVISOR_);

    __shared__ float s_sum[HS_];
    __shared__ float s_sq[HS_];
    s_sum[i] = yh;
    s_sq[i]  = yh * yh;
    __syncthreads();
    for (int s = HS_ / 2; s > 0; s >>= 1) {
        if (i < s) { s_sum[i] += s_sum[i + s]; s_sq[i] += s_sq[i + s]; }
        __syncthreads();
    }
    const float mean = s_sum[0] * (1.0f / HS_);
    const float var  = s_sq[0] * (1.0f / HS_) - mean * mean;

    float yn = (yh - mean) * __frsqrt_rn(var + 1e-5f);
    yn = yn * ln_w[h * HS_ + i] + ln_b[h * HS_ + i];

    const float g = gpre[base + i];
    const float sil = g / (1.0f + __expf(-g));
    aout[base + i] = (__bf16)(yn * sil);
}

// =====================================================================
// launch
// =====================================================================
extern "C" void kernel_launch(void* const* d_in, const int* in_sizes, int n_in,
                              void* d_out, int out_size, void* d_ws, size_t ws_size,
                              hipStream_t stream)
{
    (void)in_sizes; (void)n_in; (void)out_size; (void)ws_size;

    const float* x     = (const float*)d_in[0];
    const float* tmk   = (const float*)d_in[1];
    const float* tmv   = (const float*)d_in[2];
    const float* tmr   = (const float*)d_in[3];
    const float* tmg   = (const float*)d_in[4];
    const float* decay = (const float*)d_in[5];
    const float* faaaa = (const float*)d_in[6];
    const float* Wr    = (const float*)d_in[7];
    const float* Wk    = (const float*)d_in[8];
    const float* Wv    = (const float*)d_in[9];
    const float* Wg    = (const float*)d_in[10];
    const float* Wo    = (const float*)d_in[11];
    const float* ln_w  = (const float*)d_in[12];
    const float* ln_b  = (const float*)d_in[13];
    float* out = (float*)d_out;

    const size_t MC = (size_t)GM * GN;   // 33.5M activation elements
    const size_t CC = (size_t)GK * GN;   // 4M weight elements

    char* ws = (char*)d_ws;
    size_t off = 0;
    auto carve = [&](size_t bytes) -> void* {
        void* p = ws + off;
        off += (bytes + 255) & ~(size_t)255;
        return p;
    };

    __bf16* xk = (__bf16*)carve(MC * 2);
    __bf16* xv = (__bf16*)carve(MC * 2);
    __bf16* xr = (__bf16*)carve(MC * 2);
    __bf16* xg = (__bf16*)carve(MC * 2);
    __bf16* Wr_h = (__bf16*)carve(CC * 2);
    __bf16* Wk_h = (__bf16*)carve(CC * 2);
    __bf16* Wv_h = (__bf16*)carve(CC * 2);
    __bf16* Wg_h = (__bf16*)carve(CC * 2);
    __bf16* Wo_h = (__bf16*)carve(CC * 2);
    float* rb   = (float*)carve(MC * 4);
    float* kb   = (float*)carve(MC * 4);
    float* vb   = (float*)carve(MC * 4);
    float* gb   = (float*)carve(MC * 4);
    float* yb   = (float*)carve(MC * 4);
    __bf16* ab  = (__bf16*)carve(MC * 2);

    // 1) token-shift lerps -> bf16
    prep_kernel<<<dim3((unsigned)(MC / 256)), dim3(256), 0, stream>>>(
        x, tmk, tmv, tmr, tmg, xk, xv, xr, xg);

    // 2) weights -> bf16
    const int wgrid = (int)(CC / 256);
    f32_to_bf16_kernel<<<wgrid, 256, 0, stream>>>(Wr, Wr_h, (int)CC);
    f32_to_bf16_kernel<<<wgrid, 256, 0, stream>>>(Wk, Wk_h, (int)CC);
    f32_to_bf16_kernel<<<wgrid, 256, 0, stream>>>(Wv, Wv_h, (int)CC);
    f32_to_bf16_kernel<<<wgrid, 256, 0, stream>>>(Wg, Wg_h, (int)CC);
    f32_to_bf16_kernel<<<wgrid, 256, 0, stream>>>(Wo, Wo_h, (int)CC);

    // 3) projections r, k, v, g_pre (WMMA GEMMs)
    dim3 ggrid(GN / BN, GM / BM);   // (16, 128)
    gemm_bf16_wmma<<<ggrid, 256, 0, stream>>>(xr, Wr_h, rb);
    gemm_bf16_wmma<<<ggrid, 256, 0, stream>>>(xk, Wk_h, kb);
    gemm_bf16_wmma<<<ggrid, 256, 0, stream>>>(xv, Wv_h, vb);
    gemm_bf16_wmma<<<ggrid, 256, 0, stream>>>(xg, Wg_h, gb);

    // 4) WKV5 recurrence
    wkv5_kernel<<<B_ * H_, HS_, 0, stream>>>(rb, kb, vb, decay, faaaa, yb);

    // 5) groupnorm + SiLU gate -> bf16
    norm_gate_kernel<<<B_ * T_ * H_, HS_, 0, stream>>>(yb, gb, ln_w, ln_b, ab);

    // 6) output projection -> f32 result
    gemm_bf16_wmma<<<ggrid, 256, 0, stream>>>(ab, Wo_h, out);
}
